// FusedLinearCrossEntropy_72584947302840
// MI455X (gfx1250) — matrix operand order repaired and checked by reading, not compiled, not run on hardware
//
#include <hip/hip_runtime.h>
#include <hip/hip_bf16.h>

typedef __attribute__((ext_vector_type(16))) __bf16 v16bf;
typedef __attribute__((ext_vector_type(8)))  __bf16 v8bf;
typedef __attribute__((ext_vector_type(8)))  float  v8f;
typedef __attribute__((ext_vector_type(4)))  int    v4i;

#define B_    4
#define S_    2048
#define H_    2048
#define V_    131072
#define NTOK  (B_*(S_-1))      /* 8188 */
#define NP    8192             /* padded tokens */
#define TM    128              /* token tile   */
#define TV    256              /* vocab tile   */
#define KB    64               /* K per stage  */
#define NK    (H_/KB)          /* 32 stages    */
#define NVT   (V_/TV)          /* 512 vocab tiles */
#define LDA   (KB+8)
#define LDB   (KB+8)
#define IGN   (-100)

#if defined(__AMDGCN__) && __has_builtin(__builtin_amdgcn_global_load_async_to_lds_b128)
#define USE_ASYNC 1
#else
#define USE_ASYNC 0
#endif

#if USE_ASYNC
typedef __attribute__((address_space(1))) v4i gv4i_t;   // global int4
typedef __attribute__((address_space(3))) v4i lv4i_t;   // LDS int4
#endif

__device__ __forceinline__ void wait_async_then_noop() {
#if USE_ASYNC
#if __has_builtin(__builtin_amdgcn_s_wait_asynccnt)
  __builtin_amdgcn_s_wait_asynccnt(0);
#else
  asm volatile("s_wait_asynccnt 0" ::: "memory");
#endif
#endif
}

// ---------------- hidden fp32 -> bf16 (shifted, padded) ----------------
__global__ __launch_bounds__(256) void k_convert(const float* __restrict__ hs,
                                                 __bf16* __restrict__ hb) {
  size_t idx = ((size_t)blockIdx.x * 256 + threadIdx.x) * 8;
  size_t row = idx >> 11;              // / H_
  int    col = (int)(idx & (H_ - 1));
  __bf16 out[8];
  if (row < NTOK) {
    int b = (int)(row / (S_ - 1));
    int s = (int)(row % (S_ - 1));
    const float4* p4 = (const float4*)(hs + ((size_t)b * S_ + s) * H_ + col);
    float4 f0 = p4[0], f1 = p4[1];
    out[0]=(__bf16)f0.x; out[1]=(__bf16)f0.y; out[2]=(__bf16)f0.z; out[3]=(__bf16)f0.w;
    out[4]=(__bf16)f1.x; out[5]=(__bf16)f1.y; out[6]=(__bf16)f1.z; out[7]=(__bf16)f1.w;
  } else {
    #pragma unroll
    for (int i = 0; i < 8; ++i) out[i] = (__bf16)0.0f;
  }
  *reinterpret_cast<uint4*>(hb + row * (size_t)H_ + col) = *reinterpret_cast<uint4*>(out);
}

// ---------------- main fused GEMM + online softmax partials ----------------
// 8 waves: 2 row-groups (wr) x 4 col-groups (wc); each wave: 4 m-tiles x 4 n-tiles.
__global__ __launch_bounds__(256) void k_main(const __bf16* __restrict__ hb,
                                              const float*  __restrict__ w,
                                              const float*  __restrict__ bias,
                                              const int*    __restrict__ labels,
                                              float2*       __restrict__ partials,
                                              float*        __restrict__ targ) {
  __shared__ __bf16 shA[2][TM * LDA];
  __shared__ __bf16 shB[2][TV * LDB];
  __shared__ float  redM[4][TM];
  __shared__ float  redS[4][TM];
  __shared__ int    shLab[TM];

  const int t      = threadIdx.x;
  const int lane   = t & 31;
  const int wave   = t >> 5;
  const int wr     = wave & 1;   // 2 row-groups of 64 rows
  const int wc     = wave >> 1;  // 4 col-groups of 64 cols
  const int token0 = blockIdx.x * TM;
  const int v0     = blockIdx.y * TV;

  if (t < TM) {
    int token = token0 + t;
    int lab = -1;
    if (token < NTOK) {
      int b = token / (S_ - 1), s = token % (S_ - 1);
      lab = labels[b * S_ + s + 1];
    }
    shLab[t] = lab;
  }

  auto loadStage = [&](int st, int k0) {
    // A tile: 128 rows x 64 bf16 (already bf16 in ws) -- async DMA to LDS when available
    {
      int row = t >> 1, half = t & 1;
      const __bf16* gsrc = hb + (size_t)(token0 + row) * H_ + k0 + half * 32;
      __bf16* ldst = &shA[st][row * LDA + half * 32];
#if USE_ASYNC
      gv4i_t* g = (gv4i_t*)gsrc;
      lv4i_t* l = (lv4i_t*)ldst;
      #pragma unroll
      for (int i = 0; i < 4; ++i)
        __builtin_amdgcn_global_load_async_to_lds_b128(g + i, l + i, 0, 0);
#else
      const uint4* src = (const uint4*)gsrc;
      uint4* dst = (uint4*)ldst;
      #pragma unroll
      for (int i = 0; i < 4; ++i) dst[i] = src[i];
#endif
    }
    // B tile: 256 weight rows x 64 fp32 -> bf16 (convert in flight)
    {
      const float4* src = (const float4*)(w + (size_t)(v0 + t) * H_ + k0);
      if (k0 + 2 * KB < H_)
        __builtin_prefetch(w + (size_t)(v0 + t) * H_ + k0 + 2 * KB, 0, 0);
      __bf16* dst = &shB[st][t * LDB];
      #pragma unroll
      for (int i = 0; i < 8; ++i) {
        float4 f0 = src[2 * i], f1 = src[2 * i + 1];
        __bf16 tmp[8] = {(__bf16)f0.x, (__bf16)f0.y, (__bf16)f0.z, (__bf16)f0.w,
                         (__bf16)f1.x, (__bf16)f1.y, (__bf16)f1.z, (__bf16)f1.w};
        *(uint4*)(dst + i * 8) = *(uint4*)tmp;
      }
    }
  };

  const v8f vzero = {0.f, 0.f, 0.f, 0.f, 0.f, 0.f, 0.f, 0.f};
  v8f acc[4][4];
  #pragma unroll
  for (int mi = 0; mi < 4; ++mi)
    #pragma unroll
    for (int j = 0; j < 4; ++j) acc[mi][j] = vzero;

  loadStage(0, 0);
  wait_async_then_noop();
  __syncthreads();

  int st = 0;
  for (int ks = 0; ks < NK; ++ks) {
    if (ks + 1 < NK) loadStage(st ^ 1, (ks + 1) * KB);
    #pragma unroll
    for (int kk = 0; kk < KB; kk += 32) {
      v16bf bf[4];
      #pragma unroll
      for (int j = 0; j < 4; ++j) {
        int n  = wc * 64 + j * 16 + (lane & 15);
        int kb = kk + ((lane & 16) ? 16 : 0);
        const __bf16* pb = &shB[st][n * LDB + kb];
        v8bf b0 = *(const v8bf*)pb;        // K = kb .. kb+7
        v8bf b1 = *(const v8bf*)(pb + 8);  // K = kb+8 .. kb+15
        bf[j] = __builtin_shufflevector(b0, b1, 0,1,2,3,4,5,6,7,8,9,10,11,12,13,14,15);
      }
      v16bf af[4];
      #pragma unroll
      for (int mi = 0; mi < 4; ++mi) {
        int mr = wr * 64 + mi * 16 + (lane & 15);
        int kb = kk + ((lane & 16) ? 16 : 0);
        const __bf16* pa = &shA[st][mr * LDA + kb];
        v8bf a0 = *(const v8bf*)pa;         // K = kb .. kb+7
        v8bf a1 = *(const v8bf*)(pa + 8);   // K = kb+8 .. kb+15
        af[mi] = __builtin_shufflevector(a0, a1, 0,1,2,3,4,5,6,7,8,9,10,11,12,13,14,15);
      }
      #pragma unroll
      for (int mi = 0; mi < 4; ++mi)
        #pragma unroll
        for (int j = 0; j < 4; ++j)
          acc[mi][j] = __builtin_amdgcn_wmma_f32_16x16x32_bf16(
              false, af[mi], false, bf[j], (short)0, acc[mi][j], false, false);
    }
    wait_async_then_noop();
    __syncthreads();
    st ^= 1;
  }

  // ----- epilogue: bias, target logit, per-row max & sum(exp) -----
  float biasv[4];
  #pragma unroll
  for (int j = 0; j < 4; ++j)
    biasv[j] = bias[v0 + wc * 64 + j * 16 + (lane & 15)];

  #pragma unroll
  for (int mi = 0; mi < 4; ++mi) {
    int rbase = wr * 64 + mi * 16;
    #pragma unroll
    for (int r = 0; r < 8; ++r) {
      int row = rbase + r + ((lane & 16) ? 8 : 0);
      int lab = shLab[row];
      float vals[4];
      float mmax = -3.4e38f;
      #pragma unroll
      for (int j = 0; j < 4; ++j) {
        vals[j] = acc[mi][j][r] + biasv[j];
        mmax = fmaxf(mmax, vals[j]);
      }
      #pragma unroll
      for (int j = 0; j < 4; ++j) {
        int colg = v0 + wc * 64 + j * 16 + (lane & 15);
        if (colg == lab) targ[token0 + row] = vals[j];
      }
      #pragma unroll
      for (int off = 1; off < 16; off <<= 1)
        mmax = fmaxf(mmax, __shfl_xor(mmax, off, 32));
      float ssum = 0.f;
      #pragma unroll
      for (int j = 0; j < 4; ++j) ssum += __expf(vals[j] - mmax);
      #pragma unroll
      for (int off = 1; off < 16; off <<= 1)
        ssum += __shfl_xor(ssum, off, 32);
      if ((lane & 15) == 0) { redM[wc][row] = mmax; redS[wc][row] = ssum; }
    }
  }
  __syncthreads();

  if (t < TM) {
    float m = redM[0][t];
    #pragma unroll
    for (int c = 1; c < 4; ++c) m = fmaxf(m, redM[c][t]);
    float ss = 0.f;
    #pragma unroll
    for (int c = 0; c < 4; ++c) ss += redS[c][t] * __expf(redM[c][t] - m);
    partials[(size_t)blockIdx.y * NP + token0 + t] = make_float2(m, ss);
  }
}

// ---------------- per-token merge + deterministic block reduction ----------------
__global__ __launch_bounds__(256) void k_reduce(const float2* __restrict__ partials,
                                                const float*  __restrict__ targ,
                                                const int*    __restrict__ labels,
                                                float2*       __restrict__ blockSums) {
  __shared__ float sl[8], sc[8];
  int n = blockIdx.x * 256 + threadIdx.x;
  float loss = 0.f, cnt = 0.f;
  if (n < NTOK) {
    float m = -3.4e38f, s = 0.f;
    for (int v = 0; v < NVT; ++v) {
      float2 p = partials[(size_t)v * NP + n];
      float nm = fmaxf(m, p.x);
      s = s * __expf(m - nm) + p.y * __expf(p.x - nm);
      m = nm;
    }
    int b = n / (S_ - 1), r = n % (S_ - 1);
    int lab = labels[b * S_ + r + 1];
    if (lab != IGN) {
      loss = m + __logf(s) - targ[n];
      cnt  = 1.f;
    }
  }
  int lane = threadIdx.x & 31, wave = threadIdx.x >> 5;
  #pragma unroll
  for (int off = 1; off < 32; off <<= 1) {
    loss += __shfl_xor(loss, off, 32);
    cnt  += __shfl_xor(cnt,  off, 32);
  }
  if (lane == 0) { sl[wave] = loss; sc[wave] = cnt; }
  __syncthreads();
  if (threadIdx.x == 0) {
    float L = 0.f, C = 0.f;
    #pragma unroll
    for (int i = 0; i < 8; ++i) { L += sl[i]; C += sc[i]; }
    blockSums[blockIdx.x] = make_float2(L, C);
  }
}

__global__ void k_finalize(const float2* __restrict__ blockSums, float* __restrict__ out,
                           int nblk) {
  float L = 0.f, C = 0.f;
  for (int i = 0; i < nblk; ++i) { L += blockSums[i].x; C += blockSums[i].y; }
  out[0] = L / fmaxf(C, 1.f);
}

extern "C" void kernel_launch(void* const* d_in, const int* in_sizes, int n_in,
                              void* d_out, int out_size, void* d_ws, size_t ws_size,
                              hipStream_t stream) {
  const float* hs     = (const float*)d_in[0];
  const int*   labels = (const int*)  d_in[1];
  const float* w      = (const float*)d_in[2];
  const float* bias   = (const float*)d_in[3];
  float*       out    = (float*)d_out;

  char* wsb = (char*)d_ws;
  const size_t hbBytes   = (size_t)NP * H_ * 2;   // 33.5 MB bf16 hidden
  const size_t partBytes = (size_t)NVT * NP * 8;  // 33.5 MB float2 partials
  __bf16* hb        = (__bf16*)wsb;
  float2* partials  = (float2*)(wsb + hbBytes);
  float*  targ      = (float*)(wsb + hbBytes + partBytes);
  float2* blockSums = (float2*)(wsb + hbBytes + partBytes + (size_t)NP * 4);

  const int nRedBlk = (NTOK + 255) / 256;  // 32

  k_convert<<<(NP * (size_t)H_) / (256 * 8), 256, 0, stream>>>(hs, hb);
  dim3 grid(NP / TM, NVT);                 // token tiles fastest -> weight tiles reused from L2
  k_main<<<grid, 256, 0, stream>>>(hb, w, bias, labels, partials, targ);
  k_reduce<<<nRedBlk, 256, 0, stream>>>(partials, targ, labels, blockSums);
  k_finalize<<<1, 1, 0, stream>>>(blockSums, out, nRedBlk);
}
